// MultiHeadAttention_27702539059499
// MI455X (gfx1250) — matrix operand order repaired and checked by reading, not compiled, not run on hardware
//
#include <hip/hip_runtime.h>
#include <hip/hip_bf16.h>

typedef __attribute__((ext_vector_type(16))) _Float16 v16h;
typedef __attribute__((ext_vector_type(8)))  _Float16 v8h;
typedef __attribute__((ext_vector_type(4)))  _Float16 v4h;
typedef __attribute__((ext_vector_type(8)))  float    v8f;
typedef __attribute__((ext_vector_type(4)))  unsigned int u32x4;
typedef __attribute__((ext_vector_type(8)))  int i32x8;
typedef __attribute__((ext_vector_type(4)))  int i32x4;

#define D_MODEL   1024
#define NUM_HEADS 16
#define D_K       64
#define BATCH     4
#define SEQ       2048
#define M_TOTAL   (BATCH * SEQ)   // 8192 rows of activations

#define TILE_LD   88              // LDS row stride (halves): 176B, 16B-aligned,
                                  // bank map 44*n mod 64 is conflict-free

#if __has_builtin(__builtin_amdgcn_tensor_load_to_lds)
#define HAVE_TDM 1
#else
#define HAVE_TDM 0
#endif

// ---------------------------------------------------------------------------
// WMMA helper: D = A(16x32 f16) x B(32x16 f16) + C(16x16 f32)
// ---------------------------------------------------------------------------
__device__ __forceinline__ v8f wmma16(v16h a, v16h b, v8f c) {
  return __builtin_amdgcn_wmma_f32_16x16x32_f16(
      false, a, false, b, (short)0, c, false, false);
}

// A fragment (16x32 f16): lanes 0-15 -> row=lane, V0-3=K0..7, V4-7=K16..23;
// lanes 16-31 -> K8..15 / K24..31.
__device__ __forceinline__ v16h load_frag_a(const _Float16* p0, int ld, int k0) {
  const int lane = threadIdx.x & 31;
  const int r = lane & 15, hf = lane >> 4;
  const _Float16* p = p0 + (size_t)r * ld + k0 + hf * 8;
  v8h lo = *(const v8h*)p;
  v8h hi = *(const v8h*)(p + 16);
  v16h f;
#pragma unroll
  for (int i = 0; i < 8; ++i) { f[i] = lo[i]; f[i + 8] = hi[i]; }
  return f;
}

// B fragment (32x16 f16): lanes 0-15 -> col n=lane, V0-3=K0..7, V4-7=K16..23;
// lanes 16-31 -> K8..15 / K24..31. Source rows indexed by B column, contiguous K.
__device__ __forceinline__ v16h load_frag_b(const _Float16* p0, int ld, int k0) {
  const int lane = threadIdx.x & 31;
  const int n = lane & 15, hf = lane >> 4;
  const _Float16* p = p0 + (size_t)n * ld + k0 + hf * 8;
  v8h lo = *(const v8h*)p;
  v8h hi = *(const v8h*)(p + 16);
  v16h f;
#pragma unroll
  for (int i = 0; i < 8; ++i) { f[i] = lo[i]; f[i + 8] = hi[i]; }
  return f;
}

// ---------------------------------------------------------------------------
// TDM: 2-D tile load global->LDS, 64 rows x 64 halves, row padded 32->44
// DWORDs in LDS (pad codes 4/11) -> row stride TILE_LD halves.
// ---------------------------------------------------------------------------
#if HAVE_TDM
__device__ __forceinline__ void tdm_load_64x64(unsigned lds_addr,
                                               const void* gaddr,
                                               unsigned stride_elems) {
  const unsigned long long ga = (unsigned long long)gaddr;
  u32x4 g0;
  g0[0] = 1u;                                           // count=1, user mode
  g0[1] = lds_addr;                                     // LDS byte offset
  g0[2] = (unsigned)(ga & 0xFFFFFFFFu);                 // global_addr[31:0]
  g0[3] = (unsigned)((ga >> 32) & 0x01FFFFFFu) | (2u << 30);  // type=2
  i32x8 g1;
  g1[0] = (int)((1u << 16)                              // data_size = 2 bytes
                | (1u << 20)                            // pad_enable
                | (4u << 22)                            // pad every 32 DWORDs
                | (11u << 25));                         // pad 12 DWORDs
  g1[1] = (int)(64u << 16);                             // tensor_dim0 = 64
  g1[2] = (int)(64u << 16);                             // tensor_dim1 = 64
  g1[3] = (int)(64u << 16);                             // tile_dim0 = 64
  g1[4] = (int)64;                                      // tile_dim1 = 64
  g1[5] = (int)stride_elems;                            // tensor_dim0_stride
  g1[6] = 0;
  g1[7] = 0;
  const i32x4 gz = {0, 0, 0, 0};
#if defined(__clang_major__) && __clang_major__ >= 23
  const i32x8 gz8 = {0, 0, 0, 0, 0, 0, 0, 0};
  __builtin_amdgcn_tensor_load_to_lds(g0, g1, gz, gz, gz8, 0);
#else
  __builtin_amdgcn_tensor_load_to_lds(g0, g1, gz, gz, 0);
#endif
}
#else
// Fallback: cooperative copy of a 64 x 64-half tile into padded LDS.
template <int NTHREADS>
__device__ __forceinline__ void copy_tile_fallback(_Float16* dst,
                                                   const _Float16* src,
                                                   int src_ld) {
  const int tid = threadIdx.x;
#pragma unroll
  for (int t = 0; t < 512 / NTHREADS; ++t) {
    const int idx = tid + t * NTHREADS;  // 512 chunks of 8 halves
    const int row = idx >> 3, c = (idx & 7) * 8;
    *(v8h*)(dst + row * TILE_LD + c) =
        *(const v8h*)(src + (size_t)row * src_ld + c);
  }
}
#endif

// ---------------------------------------------------------------------------
// fp32 -> f16 conversion, 4 elements per thread-iteration
// ---------------------------------------------------------------------------
__global__ void cvt_kernel(const float* __restrict__ in,
                           _Float16* __restrict__ out, int n4) {
  int i = blockIdx.x * blockDim.x + threadIdx.x;
  const int stride = gridDim.x * blockDim.x;
  for (; i < n4; i += stride) {
    const float4 f = ((const float4*)in)[i];
    v4h h;
    h[0] = (_Float16)f.x; h[1] = (_Float16)f.y;
    h[2] = (_Float16)f.z; h[3] = (_Float16)f.w;
    ((v4h*)out)[i] = h;
  }
}

// ---------------------------------------------------------------------------
// GEMM: out[m][n] = sum_k A[m][k] * W[n][k] + bias[n]   (A @ W^T + b)
// W tile (64 cols x 64 k) TDM-staged into LDS, double-buffered, shared by all
// 8 waves. MODE 0: f16 [B,H,S,Dk]  MODE 1: f16 [B,H,Dk,S]  MODE 2: f32 [M,N]
// ---------------------------------------------------------------------------
template <int MODE>
__global__ __launch_bounds__(256) void proj_kernel(
    const _Float16* __restrict__ A, const _Float16* __restrict__ W,
    const float* __restrict__ bias, void* __restrict__ outp) {
  __shared__ _Float16 ldsW[2][64 * TILE_LD];
  const int w = threadIdx.x >> 5;
  const int lane = threadIdx.x & 31;
  const int n = lane & 15, hf = lane >> 4;
  const int row0 = blockIdx.y * 128 + w * 16;
  const int n0 = blockIdx.x * 64;
  const _Float16* Arow = A + (size_t)row0 * D_MODEL;
  const _Float16* Wt = W + (size_t)n0 * D_MODEL;

#if HAVE_TDM
  if (w == 0) tdm_load_64x64((unsigned)(uintptr_t)&ldsW[0][0], Wt, D_MODEL);
#endif

  v8f c[4] = {};
  for (int k0 = 0; k0 < D_MODEL; k0 += 64) {
    const int cur = (k0 >> 6) & 1;
#if HAVE_TDM
    if (w == 0) {
      if (k0 + 64 < D_MODEL) {
        tdm_load_64x64((unsigned)(uintptr_t)&ldsW[cur ^ 1][0], Wt + k0 + 64,
                       D_MODEL);
        __builtin_amdgcn_s_wait_tensorcnt(1);   // current tile complete
      } else {
        __builtin_amdgcn_s_wait_tensorcnt(0);
      }
    }
    __syncthreads();
#else
    __syncthreads();
    copy_tile_fallback<256>(&ldsW[cur][0], Wt + k0, D_MODEL);
    __syncthreads();
#endif
    const _Float16* Wb = &ldsW[cur][0];
    const v16h a0 = load_frag_a(Arow, D_MODEL, k0);
    const v16h a1 = load_frag_a(Arow, D_MODEL, k0 + 32);
#pragma unroll
    for (int t = 0; t < 4; ++t) {
      c[t] = wmma16(a0, load_frag_b(Wb + (t * 16) * TILE_LD, TILE_LD, 0), c[t]);
      c[t] = wmma16(a1, load_frag_b(Wb + (t * 16) * TILE_LD, TILE_LD, 32), c[t]);
    }
    __syncthreads();   // all reads done before buffer reuse
  }

#pragma unroll
  for (int t = 0; t < 4; ++t) {
    const int nc = n0 + t * 16 + n;
    const float bv = bias[nc];
#pragma unroll
    for (int i = 0; i < 8; ++i) {
      const int m = row0 + hf * 8 + i;
      const float val = c[t][i] + bv;
      if constexpr (MODE == 2) {
        ((float*)outp)[(size_t)m * D_MODEL + nc] = val;
      } else {
        const int bb = m >> 11, s = m & (SEQ - 1);
        const int h = nc >> 6, d = nc & (D_K - 1);
        const size_t idx =
            (MODE == 0)
                ? (((size_t)(bb * NUM_HEADS + h) * SEQ + s) * D_K + d)
                : (((size_t)(bb * NUM_HEADS + h) * D_K + d) * SEQ + s);
        ((_Float16*)outp)[idx] = (_Float16)val;
      }
    }
  }
}

// ---------------------------------------------------------------------------
// Flash-attention. Block = 4 waves x 16 query rows = 64 queries.
// Key loop: 64 keys/iter; K and V chunks TDM-staged into LDS (double-buffered).
// ---------------------------------------------------------------------------
__global__ __launch_bounds__(128) void attn_kernel(
    const _Float16* __restrict__ Q, const _Float16* __restrict__ K,
    const _Float16* __restrict__ Vt, _Float16* __restrict__ O) {
  __shared__ _Float16 ldsK[2][64 * TILE_LD];   // keys x d (padded rows)
  __shared__ _Float16 ldsV[2][64 * TILE_LD];   // d x s    (padded rows)
  __shared__ _Float16 ldsP[4][16 * TILE_LD];   // wave-private P tiles (16x64)

  const int w = threadIdx.x >> 5;
  const int lane = threadIdx.x & 31;
  const int n = lane & 15, hf = lane >> 4;
  const int head = blockIdx.y;                 // b*NUM_HEADS + h
  const int q0 = blockIdx.x * 64 + w * 16;
  const _Float16* Qh = Q + (size_t)head * SEQ * D_K;   // [s][d]
  const _Float16* Kh = K + (size_t)head * SEQ * D_K;   // [s][d]
  const _Float16* Vh = Vt + (size_t)head * D_K * SEQ;  // [d][s]
  _Float16* myP = &ldsP[w][0];

  // Q fragments, with 1/sqrt(Dk) folded in once
  v16h qa0 = load_frag_a(Qh + (size_t)q0 * D_K, D_K, 0);
  v16h qa1 = load_frag_a(Qh + (size_t)q0 * D_K, D_K, 32);
#pragma unroll
  for (int i = 0; i < 16; ++i) {
    qa0[i] = qa0[i] * (_Float16)0.125f;
    qa1[i] = qa1[i] * (_Float16)0.125f;
  }

  v16h ones;
#pragma unroll
  for (int i = 0; i < 16; ++i) ones[i] = (_Float16)1.0f;

  float mrow[8], lrow[8], arow[8];
  v8f o[4] = {};
#pragma unroll
  for (int i = 0; i < 8; ++i) { mrow[i] = -1e30f; lrow[i] = 0.0f; }

#if HAVE_TDM
  if (w == 0) {
    tdm_load_64x64((unsigned)(uintptr_t)&ldsK[0][0], Kh, D_K);
    tdm_load_64x64((unsigned)(uintptr_t)&ldsV[0][0], Vh, SEQ);
  }
#endif

  for (int j0 = 0; j0 < SEQ; j0 += 64) {
    const int cur = (j0 >> 6) & 1;
#if HAVE_TDM
    if (w == 0) {
      if (j0 + 64 < SEQ) {
        tdm_load_64x64((unsigned)(uintptr_t)&ldsK[cur ^ 1][0],
                       Kh + (size_t)(j0 + 64) * D_K, D_K);
        tdm_load_64x64((unsigned)(uintptr_t)&ldsV[cur ^ 1][0], Vh + (j0 + 64),
                       SEQ);
        __builtin_amdgcn_s_wait_tensorcnt(2);   // current pair complete
      } else {
        __builtin_amdgcn_s_wait_tensorcnt(0);
      }
    }
    __syncthreads();
#else
    __syncthreads();
    copy_tile_fallback<128>(&ldsK[cur][0], Kh + (size_t)j0 * D_K, D_K);
    copy_tile_fallback<128>(&ldsV[cur][0], Vh + j0, SEQ);
    __syncthreads();
#endif
    const _Float16* Kb = &ldsK[cur][0];
    const _Float16* Vb = &ldsV[cur][0];

    // scores for 4 key sub-tiles (pre-scaled Q)
    v8f s[4];
#pragma unroll
    for (int c4 = 0; c4 < 4; ++c4) {
      v8f acc = {};
      acc = wmma16(qa0, load_frag_b(Kb + (c4 * 16) * TILE_LD, TILE_LD, 0), acc);
      acc = wmma16(qa1, load_frag_b(Kb + (c4 * 16) * TILE_LD, TILE_LD, 32), acc);
      s[c4] = acc;
    }

    // online softmax: one max-reduction per row per 64 keys
#pragma unroll
    for (int i = 0; i < 8; ++i) {
      float t = fmaxf(fmaxf(s[0][i], s[1][i]), fmaxf(s[2][i], s[3][i]));
      t = fmaxf(t, __shfl_xor(t, 1, 16));
      t = fmaxf(t, __shfl_xor(t, 2, 16));
      t = fmaxf(t, __shfl_xor(t, 4, 16));
      t = fmaxf(t, __shfl_xor(t, 8, 16));
      const float mn = fmaxf(mrow[i], t);
      arow[i] = __expf(mrow[i] - mn);
      mrow[i] = mn;
      const int r = hf * 8 + i;
#pragma unroll
      for (int c4 = 0; c4 < 4; ++c4) {
        const float p = __expf(s[c4][i] - mn);
        myP[r * TILE_LD + c4 * 16 + n] = (_Float16)p;   // C-layout -> LDS
        o[c4][i] *= arow[i];
      }
    }
    asm volatile("s_wait_dscnt 0" ::: "memory");  // wave-local LDS RAW fence

    // reload P as A fragments (16x64 -> two 16x32)
    const v16h pa0 = load_frag_a(myP, TILE_LD, 0);
    const v16h pa1 = load_frag_a(myP, TILE_LD, 32);

    // row sums via WMMA against all-ones B (replaces shuffle reduction)
    v8f rs = {};
    rs = wmma16(pa0, ones, rs);
    rs = wmma16(pa1, ones, rs);
#pragma unroll
    for (int i = 0; i < 8; ++i) lrow[i] = lrow[i] * arow[i] + rs[i];

    // O += P @ V
#pragma unroll
    for (int t4 = 0; t4 < 4; ++t4) {
      o[t4] = wmma16(pa0, load_frag_b(Vb + (t4 * 16) * TILE_LD, TILE_LD, 0), o[t4]);
      o[t4] = wmma16(pa1, load_frag_b(Vb + (t4 * 16) * TILE_LD, TILE_LD, 32), o[t4]);
    }
    __syncthreads();   // all reads done before buffer reuse
  }

  // normalize and store f16 attention output at [m][h*64 + d]
  const int bb = head >> 4, hh = head & (NUM_HEADS - 1);
#pragma unroll
  for (int i = 0; i < 8; ++i) {
    const float inv = 1.0f / lrow[i];
    const int m = bb * SEQ + q0 + hf * 8 + i;
#pragma unroll
    for (int t4 = 0; t4 < 4; ++t4) {
      const int col = hh * D_K + t4 * 16 + n;
      O[(size_t)m * D_MODEL + col] = (_Float16)(o[t4][i] * inv);
    }
  }
}

// ---------------------------------------------------------------------------
// Host launcher
// ---------------------------------------------------------------------------
extern "C" void kernel_launch(void* const* d_in, const int* in_sizes, int n_in,
                              void* d_out, int out_size, void* d_ws,
                              size_t ws_size, hipStream_t stream) {
  (void)in_sizes; (void)n_in; (void)out_size; (void)ws_size;
  const float* q  = (const float*)d_in[0];
  const float* k  = (const float*)d_in[1];
  const float* v  = (const float*)d_in[2];
  const float* Wq = (const float*)d_in[3];
  const float* bq = (const float*)d_in[4];
  const float* Wk = (const float*)d_in[5];
  const float* bk = (const float*)d_in[6];
  const float* Wv = (const float*)d_in[7];
  const float* bv = (const float*)d_in[8];
  const float* Wo = (const float*)d_in[9];
  const float* bo = (const float*)d_in[10];

  const size_t NACT = (size_t)M_TOTAL * D_MODEL;
  const size_t NW   = (size_t)D_MODEL * D_MODEL;
  _Float16* ws  = (_Float16*)d_ws;
  _Float16* hq  = ws;
  _Float16* hk  = hq  + NACT;
  _Float16* hv  = hk  + NACT;
  _Float16* hWq = hv  + NACT;
  _Float16* hWk = hWq + NW;
  _Float16* hWv = hWk + NW;
  _Float16* hWo = hWv + NW;
  _Float16* hQ  = hWo + NW;         // [B,H,S,Dk]
  _Float16* hK  = hQ  + NACT;       // [B,H,S,Dk]
  _Float16* hVt = hK  + NACT;       // [B,H,Dk,S]
  _Float16* hO  = hVt + NACT;       // [M, D_MODEL]

  cvt_kernel<<<1024, 256, 0, stream>>>(q,  hq,  (int)(NACT / 4));
  cvt_kernel<<<1024, 256, 0, stream>>>(k,  hk,  (int)(NACT / 4));
  cvt_kernel<<<1024, 256, 0, stream>>>(v,  hv,  (int)(NACT / 4));
  cvt_kernel<<<512,  256, 0, stream>>>(Wq, hWq, (int)(NW / 4));
  cvt_kernel<<<512,  256, 0, stream>>>(Wk, hWk, (int)(NW / 4));
  cvt_kernel<<<512,  256, 0, stream>>>(Wv, hWv, (int)(NW / 4));
  cvt_kernel<<<512,  256, 0, stream>>>(Wo, hWo, (int)(NW / 4));

  const dim3 pgrid(D_MODEL / 64, M_TOTAL / 128);
  proj_kernel<0><<<pgrid, 256, 0, stream>>>(hq, hWq, bq, (void*)hQ);
  proj_kernel<0><<<pgrid, 256, 0, stream>>>(hk, hWk, bk, (void*)hK);
  proj_kernel<1><<<pgrid, 256, 0, stream>>>(hv, hWv, bv, (void*)hVt);

  attn_kernel<<<dim3(SEQ / 64, BATCH * NUM_HEADS), 128, 0, stream>>>(hQ, hK,
                                                                     hVt, hO);

  proj_kernel<2><<<pgrid, 256, 0, stream>>>(hO, hWo, bo, d_out);
}
